// GCNModule_27401891348679
// MI455X (gfx1250) — matrix operand order repaired and checked by reading, not compiled, not run on hardware
//
#include <hip/hip_runtime.h>
#include <stdint.h>
#include <stddef.h>

// ---------------------------------------------------------------------------
// 3-layer GCN for MI455X (gfx1250, wave32).
//   Per layer:  H = X @ W              (WMMA bf16 16x16x32, f32 accumulate;
//                                       W staged into LDS by the TDM with
//                                       hardware pad for bank-conflict-free
//                                       reads)
//               agg = b + dinv^2 * H   (fused into GEMM epilogue)
//               agg[dst] += norm_e * H[src]   (fp32 HW atomics, L2-resident)
//               Xb_next = bf16(relu(agg))     (fused relu + requant)
// Scatter traffic (~1 GB/layer) dominates the roofline; GEMM is ~50 MB/layer.
// ---------------------------------------------------------------------------

typedef __bf16 v16bf __attribute__((ext_vector_type(16)));
typedef __bf16 v8bf  __attribute__((ext_vector_type(8)));
typedef float  v8f   __attribute__((ext_vector_type(8)));
typedef unsigned int u32x4 __attribute__((ext_vector_type(4)));
typedef int          i32x4 __attribute__((ext_vector_type(4)));
typedef int          i32x8 __attribute__((ext_vector_type(8)));

#define DHID 128          // D == H == 128
#define WPITCH 136        // LDS pitch (ushorts): 272B rows -> bank stride 4
#define TILES_PER_BLOCK 8

__device__ __forceinline__ unsigned short f32_to_bf16_rne(float f) {
  union { float f; unsigned u; } v; v.f = f;
  unsigned u = v.u;
  u += 0x7FFFu + ((u >> 16) & 1u);   // round-to-nearest-even
  return (unsigned short)(u >> 16);
}

__device__ __forceinline__ void atomic_fadd(float* p, float v) {
  // relaxed, device scope -> global_atomic_add_f32 (no CAS loop)
  __hip_atomic_fetch_add(p, v, __ATOMIC_RELAXED, __HIP_MEMORY_SCOPE_AGENT);
}

// ---------------- degree / normalization -----------------------------------

__global__ void k_deg_init(float* __restrict__ deg, int n) {
  int i = blockIdx.x * blockDim.x + threadIdx.x;
  if (i < n) deg[i] = 1.0f;                      // self-loop
}

__global__ void k_deg_count(const long long* __restrict__ ei,
                            float* __restrict__ deg, int E) {
  int e = blockIdx.x * blockDim.x + threadIdx.x;
  if (e < E) {
    int d = (int)ei[(size_t)E + e];              // dst row
    atomic_fadd(&deg[d], 1.0f);
  }
}

__global__ void k_dinv(float* __restrict__ deg, int n) {
  int i = blockIdx.x * blockDim.x + threadIdx.x;
  if (i < n) deg[i] = rsqrtf(deg[i]);            // deg >= 1 always
}

// pack (src, dst, norm) once; 3 scatter passes then need one broadcast b128
__global__ void k_edge_prep(const long long* __restrict__ ei,
                            const float* __restrict__ dinv,
                            int4* __restrict__ rec, int E) {
  int e = blockIdx.x * blockDim.x + threadIdx.x;
  if (e < E) {
    int s = (int)ei[e];
    int d = (int)ei[(size_t)E + e];
    float w = dinv[s] * dinv[d];
    rec[e] = make_int4(s, d, __float_as_int(w), 0);
  }
}

// ---------------- precision conversion --------------------------------------

__global__ void k_f32_to_bf16(const float* __restrict__ src,
                              unsigned short* __restrict__ dst, int n4) {
  int i = blockIdx.x * blockDim.x + threadIdx.x;
  if (i < n4) {
    float4 f = ((const float4*)src)[i];
    ushort4 o;
    o.x = f32_to_bf16_rne(f.x); o.y = f32_to_bf16_rne(f.y);
    o.z = f32_to_bf16_rne(f.z); o.w = f32_to_bf16_rne(f.w);
    ((ushort4*)dst)[i] = o;
  }
}

// W (k-major 128x128 f32) -> Wt (n-major 128x128 bf16): B-fragment K-runs
// become contiguous.
__global__ void k_w_transpose_bf16(const float* __restrict__ W,
                                   unsigned short* __restrict__ Wt) {
  int idx = blockIdx.x * blockDim.x + threadIdx.x;   // 16384
  if (idx < DHID * DHID) {
    int n = idx & (DHID - 1);
    int k = idx >> 7;
    Wt[(size_t)n * DHID + k] = f32_to_bf16_rne(W[idx]);
  }
}

// ---------------- WMMA GEMM + fused self-loop/bias epilogue -----------------
// Block = 256 threads = 8 waves; wave w owns output columns [16w, 16w+16).
// W tile (32KB bf16) is DMA'd into LDS by the Tensor Data Mover with a
// hardware pad of 4 DWORDs per 64-DWORD row -> 272B pitch (bank stride 4).

__global__ __launch_bounds__(256)
void k_gemm_wmma(const unsigned short* __restrict__ Xb,
                 const unsigned short* __restrict__ Wt,
                 const float* __restrict__ dinv,
                 const float* __restrict__ bias,
                 float* __restrict__ Hm,
                 float* __restrict__ agg,
                 int nrows) {
  __shared__ unsigned short sW[DHID * WPITCH];       // 34816 B

  // ---- TDM: stage Wt (128 rows x 256B) into padded LDS -------------------
  {
    unsigned ldsAddr = (unsigned)(uintptr_t)(&sW[0]);   // low 32 bits = LDS offset
    unsigned long long ga = (unsigned long long)(uintptr_t)Wt;
    u32x4 g0 = { 1u,                                     // count=1, no gather
                 ldsAddr,                                // lds_addr
                 (unsigned)(ga & 0xFFFFFFFFu),           // global_addr[31:0]
                 (unsigned)((ga >> 32) & 0x1FFFFFFu) | 0x80000000u }; // [56:32] | type=2
    i32x8 g1 = { (int)((2u << 16) | (1u << 20) | (5u << 22) | (3u << 25)),
                 // data_size=4B, pad_enable, pad_interval=64DW, pad_amount=4DW
                 (int)(64u  << 16),    // tensor_dim0 = 64 (x4B = 256B row)
                 (int)(128u << 16),    // tensor_dim1 = 128 rows
                 (int)(64u  << 16),    // tile_dim0   = 64
                 128,                  // tile_dim1   = 128
                 64,                   // tensor_dim0_stride = 64 (contiguous)
                 0, 0 };
    i32x4 gz4 = { 0, 0, 0, 0 };
    i32x8 gz8 = { 0, 0, 0, 0, 0, 0, 0, 0 };
    if (threadIdx.x < 32) {            // one wave issues the DMA
      __builtin_amdgcn_tensor_load_to_lds(g0, g1, gz4, gz4, gz8, 0);
      __builtin_amdgcn_s_wait_tensorcnt(0);
    }
  }
  __syncthreads();

  const int wave = threadIdx.x >> 5;
  const int lane = threadIdx.x & 31;
  const int r    = lane & 15;
  const int half = lane >> 4;
  const int n0   = wave * 16;

  const unsigned short* sB = &sW[(n0 + r) * WPITCH + half * 8];
  const float bcol = bias[n0 + r];

  for (int t = 0; t < TILES_PER_BLOCK; ++t) {
    int m0 = (blockIdx.x * TILES_PER_BLOCK + t) * 16;
    if (m0 >= nrows) return;

    // prefetch next tile's A row
    if (t + 1 < TILES_PER_BLOCK && m0 + 16 < nrows)
      __builtin_prefetch(Xb + (size_t)(m0 + 16 + r) * DHID, 0, 3);

    int arow = m0 + r; if (arow >= nrows) arow = nrows - 1;
    const unsigned short* aRow = Xb + (size_t)arow * DHID;

    v8f acc = {};
#pragma unroll
    for (int kk = 0; kk < DHID; kk += 32) {
      // A fragment: lane l holds K = half*8+0..7 and 16+half*8+0..7 of row r
      v8bf alo = *(const v8bf*)(aRow + kk + half * 8);
      v8bf ahi = *(const v8bf*)(aRow + kk + 16 + half * 8);
      v16bf a = __builtin_shufflevector(alo, ahi,
                  0,1,2,3,4,5,6,7,8,9,10,11,12,13,14,15);
      // B fragment: lane l holds K = half*8+0..7 and 16+half*8+0..7 of column n0+r
      v8bf blo = *(const v8bf*)(sB + kk);
      v8bf bhi = *(const v8bf*)(sB + kk + 16);
      v16bf b = __builtin_shufflevector(blo, bhi,
                  0,1,2,3,4,5,6,7,8,9,10,11,12,13,14,15);
      acc = __builtin_amdgcn_wmma_f32_16x16x32_bf16(
          false, a, false, b, (short)0, acc, false, false);
    }

    // D layout: VGPR j -> row m0 + j + 8*half, col n0 + r
    // Fused epilogue: Hm = acc; agg = bias + dinv^2 * acc (self-loop term)
    if (m0 + 16 <= nrows) {            // fast path: unguarded stores
#pragma unroll
      for (int j = 0; j < 8; ++j) {
        int row = m0 + j + 8 * half;
        float di = dinv[row];
        size_t o = (size_t)row * DHID + n0 + r;
        Hm[o]  = acc[j];
        agg[o] = fmaf(acc[j], di * di, bcol);
      }
    } else {                           // partial tile (unused when N%16==0)
#pragma unroll
      for (int j = 0; j < 8; ++j) {
        int row = m0 + j + 8 * half;
        if (row < nrows) {
          float di = dinv[row];
          size_t o = (size_t)row * DHID + n0 + r;
          Hm[o]  = acc[j];
          agg[o] = fmaf(acc[j], di * di, bcol);
        }
      }
    }
  }
}

// ---------------- edge scatter ----------------------------------------------
// one wave per edge: broadcast b128 record, coalesced 512B gather of H[src],
// 4 HW f32 atomics per lane into agg[dst].

__global__ void k_edge_scatter(const float* __restrict__ Hm,
                               const int4* __restrict__ rec,
                               float* __restrict__ agg, int E) {
  int t = blockIdx.x * blockDim.x + threadIdx.x;
  int e = t >> 5;
  if (e >= E) return;
  int lane = t & 31;
  int4 rc = rec[e];                    // same addr in all lanes -> broadcast
  float w = __int_as_float(rc.z);
  float4 hv = ((const float4*)(Hm + (size_t)rc.x * DHID))[lane];
  float* dstp = agg + (size_t)rc.y * DHID + lane * 4;
  atomic_fadd(dstp + 0, hv.x * w);
  atomic_fadd(dstp + 1, hv.y * w);
  atomic_fadd(dstp + 2, hv.z * w);
  atomic_fadd(dstp + 3, hv.w * w);
}

// ---------------- relu epilogues -------------------------------------------

__global__ void k_relu_to_bf16(const float* __restrict__ agg,
                               unsigned short* __restrict__ dst, int n4) {
  int i = blockIdx.x * blockDim.x + threadIdx.x;
  if (i < n4) {
    float4 f = ((const float4*)agg)[i];
    ushort4 o;
    o.x = f32_to_bf16_rne(fmaxf(f.x, 0.f)); o.y = f32_to_bf16_rne(fmaxf(f.y, 0.f));
    o.z = f32_to_bf16_rne(fmaxf(f.z, 0.f)); o.w = f32_to_bf16_rne(fmaxf(f.w, 0.f));
    ((ushort4*)dst)[i] = o;
  }
}

__global__ void k_relu_f32(const float* __restrict__ agg,
                           float* __restrict__ out, int n4) {
  int i = blockIdx.x * blockDim.x + threadIdx.x;
  if (i < n4) {
    float4 f = ((const float4*)agg)[i];
    f.x = fmaxf(f.x, 0.f); f.y = fmaxf(f.y, 0.f);
    f.z = fmaxf(f.z, 0.f); f.w = fmaxf(f.w, 0.f);
    ((float4*)out)[i] = f;
  }
}

// ---------------------------------------------------------------------------

extern "C" void kernel_launch(void* const* d_in, const int* in_sizes, int n_in,
                              void* d_out, int out_size, void* d_ws, size_t ws_size,
                              hipStream_t stream) {
  const float*     x  = (const float*)d_in[0];
  const long long* ei = (const long long*)d_in[1];   // int64 edge_index [2,E]

  const int N = in_sizes[0] / DHID;
  const int E = in_sizes[1] / 2;
  const int total = N * DHID;

  // workspace carve-out (256B aligned)
  char* ws = (char*)d_ws;
  size_t off = 0;
  auto carve = [&](size_t bytes) {
    void* p = ws + off;
    off = (off + bytes + 255) & ~(size_t)255;
    return p;
  };
  float*          dinv = (float*)carve((size_t)N * 4);             // deg -> dinv
  unsigned short* Xb   = (unsigned short*)carve((size_t)total * 2);
  unsigned short* Wt   = (unsigned short*)carve((size_t)DHID * DHID * 2);
  float*          Hm   = (float*)carve((size_t)total * 4);
  float*          agg  = (float*)carve((size_t)total * 4);
  int4*           rec  = (int4*)carve((size_t)E * sizeof(int4));
  (void)ws_size;

  const int TPB = 256;
  // normalization + edge records (once)
  k_deg_init <<<(N + TPB - 1) / TPB, TPB, 0, stream>>>(dinv, N);
  k_deg_count<<<(E + TPB - 1) / TPB, TPB, 0, stream>>>(ei, dinv, E);
  k_dinv     <<<(N + TPB - 1) / TPB, TPB, 0, stream>>>(dinv, N);
  k_edge_prep<<<(E + TPB - 1) / TPB, TPB, 0, stream>>>(ei, dinv, rec, E);

  // layer-0 input -> bf16
  k_f32_to_bf16<<<(total / 4 + TPB - 1) / TPB, TPB, 0, stream>>>(x, Xb, total / 4);

  const int rowTiles  = (N + 15) / 16;
  const int gemmBlks  = (rowTiles + TILES_PER_BLOCK - 1) / TILES_PER_BLOCK;
  const long long scatterThreads = (long long)E * 32;
  const int scatterBlks = (int)((scatterThreads + TPB - 1) / TPB);

  for (int layer = 0; layer < 3; ++layer) {
    const float* W = (const float*)d_in[2 + 2 * layer];
    const float* b = (const float*)d_in[3 + 2 * layer];

    k_w_transpose_bf16<<<(DHID * DHID + TPB - 1) / TPB, TPB, 0, stream>>>(W, Wt);
    k_gemm_wmma<<<gemmBlks, TPB, 0, stream>>>(Xb, Wt, dinv, b, Hm, agg, N);
    k_edge_scatter<<<scatterBlks, TPB, 0, stream>>>(Hm, rec, agg, E);

    if (layer < 2) {
      k_relu_to_bf16<<<(total / 4 + TPB - 1) / TPB, TPB, 0, stream>>>(agg, Xb, total / 4);
    } else {
      k_relu_f32<<<(total / 4 + TPB - 1) / TPB, TPB, 0, stream>>>(agg, (float*)d_out, total / 4);
    }
  }
}
